// MultiheadLatentAttention_91199335563654
// MI455X (gfx1250) — matrix operand order repaired and checked by reading, not compiled, not run on hardware
//
#include <hip/hip_runtime.h>
#include <hip/hip_bf16.h>

#define H_DIM 7168
#define NH    128
#define DQ    1536
#define DKV   512
#define HD    128
#define RD    64

typedef __bf16 bf16_t;
typedef __attribute__((ext_vector_type(16))) __bf16 v16bf;
typedef __attribute__((ext_vector_type(8)))  __bf16 v8bf;
typedef __attribute__((ext_vector_type(8)))  float  v8f;
typedef __attribute__((ext_vector_type(4)))  float  v4f;   // clang vector: OK for nontemporal builtins

__device__ __forceinline__ v8f wmma_bf16(v16bf a, v16bf b, v8f c) {
  return __builtin_amdgcn_wmma_f32_16x16x32_bf16(false, a, false, b, (short)0, c, false, false);
}

// Assemble a 16-element bf16 fragment from two 16-byte LDS chunks.
__device__ __forceinline__ v16bf ld_frag2(const bf16_t* p0, const bf16_t* p1) {
  v8bf lo = *(const v8bf*)p0;
  v8bf hi = *(const v8bf*)p1;
  v16bf r;
#pragma unroll
  for (int i = 0; i < 8; ++i) { r[i] = lo[i]; r[8 + i] = hi[i]; }
  return r;
}
__device__ __forceinline__ v16bf ld_frag16(const bf16_t* p) { return ld_frag2(p, p + 8); }

// ---------------------------------------------------------------------------
// GEMM: C[M,N] = A[M,K]*B[K,N] + bias[N]  (fp32 I/O, bf16 WMMA, f32 accum)
// 256 threads (8 waves), tile 128x64, k-step 32, double-buffered LDS.
// grid.x indexes M-tiles so concurrent blocks share one weight panel; weights
// are streamed non-temporally (read once), activations stay L2-resident.
// ---------------------------------------------------------------------------
#define BM 128
#define BN 64
#define BK 32

__global__ __launch_bounds__(256) void gemm_bf16(
    const float* __restrict__ A, const float* __restrict__ B,
    const float* __restrict__ bias, float* __restrict__ C,
    int M, int N, int K)
{
  __shared__ __align__(64) bf16_t As[2][BM][BK];   // row-major, 64B rows
  __shared__ __align__(64) bf16_t Bt[2][BN][BK];   // transposed: Bt[n][k]

  const int t    = threadIdx.x;
  const int wave = t >> 5;           // 0..7 -> 16 rows each
  const int lane = t & 31;
  const int hi   = lane >> 4;
  const int l16  = lane & 15;
  const int m0   = blockIdx.x * BM;
  const int n0   = blockIdx.y * BN;

  // staging coordinates
  const int tr = t >> 3;             // 0..31 (A rows tr + 32*i)
  const int tc = (t & 7) * 4;        // 0..28
  const int bk = t >> 4;             // 0..15 (B rows bk + 16*i)
  const int bn = (t & 15) * 4;       // 0..60

  v4f ra[4], rb[2];

  auto stage_regs = [&](int k0) {
#pragma unroll
    for (int i = 0; i < 4; ++i)
      ra[i] = *(const v4f*)(A + (size_t)(m0 + tr + 32 * i) * K + k0 + tc);
#pragma unroll
    for (int i = 0; i < 2; ++i)
      rb[i] = __builtin_nontemporal_load(
          (const v4f*)(B + (size_t)(k0 + bk + 16 * i) * N + n0 + bn));
  };
  auto stage_lds = [&](int buf) {
#pragma unroll
    for (int i = 0; i < 4; ++i) {
      bf16_t* p = &As[buf][tr + 32 * i][tc];
      p[0] = (bf16_t)ra[i].x; p[1] = (bf16_t)ra[i].y;
      p[2] = (bf16_t)ra[i].z; p[3] = (bf16_t)ra[i].w;
    }
#pragma unroll
    for (int i = 0; i < 2; ++i) {
      const int kk = bk + 16 * i;
      Bt[buf][bn + 0][kk] = (bf16_t)rb[i].x;
      Bt[buf][bn + 1][kk] = (bf16_t)rb[i].y;
      Bt[buf][bn + 2][kk] = (bf16_t)rb[i].z;
      Bt[buf][bn + 3][kk] = (bf16_t)rb[i].w;
    }
  };

  v8f acc[4];
#pragma unroll
  for (int c = 0; c < 4; ++c)
#pragma unroll
    for (int i = 0; i < 8; ++i) acc[c][i] = 0.0f;

  const int nk = K / BK;
  stage_regs(0);
  stage_lds(0);
  for (int kb = 0; kb < nk; ++kb) {
    if (kb + 1 < nk) stage_regs((kb + 1) * BK);   // global loads in flight
    __syncthreads();                              // buf[kb&1] ready for all
    const int buf = kb & 1;
    // batch all fragment loads, then issue WMMAs back-to-back
    const bf16_t* arow = &As[buf][wave * 16 + l16][0];
    const v16bf a = ld_frag2(arow + (hi ? 8 : 0), arow + (hi ? 24 : 16));
    v16bf bf[4];
#pragma unroll
    for (int c = 0; c < 4; ++c)
      bf[c] = ld_frag2(&Bt[buf][c * 16 + l16][hi * 8], &Bt[buf][c * 16 + l16][16 + hi * 8]);
#pragma unroll
    for (int c = 0; c < 4; ++c)
      acc[c] = wmma_bf16(a, bf[c], acc[c]);
    if (kb + 1 < nk) stage_lds((kb + 1) & 1);     // fill other buffer
  }

  // Epilogue: lane holds col = l16, rows r + hi*8
#pragma unroll
  for (int c = 0; c < 4; ++c) {
    const int col = n0 + c * 16 + l16;
    const float bb = bias ? bias[col] : 0.0f;
#pragma unroll
    for (int r = 0; r < 8; ++r) {
      const int row = m0 + wave * 16 + hi * 8 + r;
      C[(size_t)row * N + col] = acc[c][r] + bb;
    }
  }
}

// ---------------------------------------------------------------------------
// RoPE in-place on x[s][h*RD + d], half-rotation, RD=64.
// ---------------------------------------------------------------------------
__global__ void rope_kernel(float* __restrict__ x, int S, int nh, int ld)
{
  const int idx = blockIdx.x * blockDim.x + threadIdx.x;
  const int total = S * nh * 32;
  if (idx >= total) return;
  const int i = idx & 31;
  const int h = (idx >> 5) % nh;
  const int s = idx / (32 * nh);
  const float inv_freq = __powf(10000.0f, -(2.0f * (float)i) / 64.0f);
  const float ang = (float)s * inv_freq;
  const float c = __cosf(ang), sn = __sinf(ang);
  float* p = x + (size_t)s * ld + h * RD;
  const float x1 = p[i], x2 = p[i + 32];
  p[i]      = x1 * c - x2 * sn;
  p[i + 32] = x2 * c + x1 * sn;
}

// ---------------------------------------------------------------------------
// Flash attention: one block per (64 q-rows, head); 4 waves, 16 q-rows each.
// q = [q_c | q_r] (d=192, pre-scaled); kv streamed 32 rows/step.
// ---------------------------------------------------------------------------
__global__ __launch_bounds__(128) void attn_kernel(
    const float* __restrict__ q_c, const float* __restrict__ q_r,
    const float* __restrict__ kv_c, const float* __restrict__ k_r,
    float* __restrict__ o, int S)
{
  __shared__ __align__(64) bf16_t Kl[32][200];    // [kv_rel][d] (400B rows)
  __shared__ __align__(64) bf16_t Vt[128][40];    // [d][kv_rel] (80B rows)
  __shared__ __align__(64) bf16_t Pl[4][16][32];  // per-wave P tile

  const int q0   = blockIdx.x * 64;
  const int hh   = blockIdx.y;
  const int t    = threadIdx.x;
  const int wave = t >> 5;
  const int lane = t & 31;
  const int hi   = lane >> 4;
  const int l16  = lane & 15;
  const float scale = 0.07216878364870322f;       // 1/sqrt(192)

  // Preload q fragments for this wave's 16 rows: 6 x (16x32) bf16 A-frags
  v16bf qf[6];
  {
    const int qrow = q0 + wave * 16 + l16;
    const int o0 = hi ? 8 : 0, o1 = hi ? 24 : 16;
#pragma unroll
    for (int f = 0; f < 6; ++f) {
      const float* base = (f < 4)
          ? q_c + (size_t)qrow * (NH * HD) + hh * HD + f * 32
          : q_r + (size_t)qrow * (NH * RD) + hh * RD + (f - 4) * 32;
#pragma unroll
      for (int i = 0; i < 8; ++i) {
        qf[f][i]     = (bf16_t)(base[o0 + i] * scale);
        qf[f][8 + i] = (bf16_t)(base[o1 + i] * scale);
      }
    }
  }

  float mrow[8], lrow[8];
  v8f oacc[8];
#pragma unroll
  for (int r = 0; r < 8; ++r) { mrow[r] = -1e30f; lrow[r] = 0.0f; }
#pragma unroll
  for (int f = 0; f < 8; ++f)
#pragma unroll
    for (int r = 0; r < 8; ++r) oacc[f][r] = 0.0f;

  const int sr = t >> 2;             // 0..31: staged kv row
  const int sq = t & 3;              // quarter

  for (int kv0 = 0; kv0 < S; kv0 += 32) {
    // ---- batched register staging (one loadcnt wait for all) ----
    const float* kc = kv_c + (size_t)(kv0 + sr) * (NH * 2 * HD) + hh * 2 * HD;
    const float* kr = k_r + (size_t)(kv0 + sr) * RD;
    const float* vv = kc + HD;
    v4f kreg[12], vreg[8];
#pragma unroll
    for (int j = 0; j < 12; ++j) {
      const int gi = sq * 12 + j;    // float4 index in 48-f4 row (32 kc + 16 kr)
      kreg[j] = (gi < 32) ? *(const v4f*)(kc + gi * 4)
                          : *(const v4f*)(kr + (gi - 32) * 4);
    }
#pragma unroll
    for (int j = 0; j < 8; ++j)
      vreg[j] = *(const v4f*)(vv + sq * 32 + 4 * j);

    // ---- convert + LDS store ----
#pragma unroll
    for (int j = 0; j < 12; ++j) {
      bf16_t* p = &Kl[sr][(sq * 12 + j) * 4];
      p[0] = (bf16_t)kreg[j].x; p[1] = (bf16_t)kreg[j].y;
      p[2] = (bf16_t)kreg[j].z; p[3] = (bf16_t)kreg[j].w;
    }
#pragma unroll
    for (int j = 0; j < 8; ++j) {
      const int d0 = sq * 32 + 4 * j;
      Vt[d0 + 0][sr] = (bf16_t)vreg[j].x;
      Vt[d0 + 1][sr] = (bf16_t)vreg[j].y;
      Vt[d0 + 2][sr] = (bf16_t)vreg[j].z;
      Vt[d0 + 3][sr] = (bf16_t)vreg[j].w;
    }
    __syncthreads();

    // ---- scores: two 16x16 tiles over 32 kv columns, K=192 in 6 chunks ----
    v8f sc[2];
#pragma unroll
    for (int kt = 0; kt < 2; ++kt) {
#pragma unroll
      for (int i = 0; i < 8; ++i) sc[kt][i] = 0.0f;
      v16bf kb[6];
#pragma unroll
      for (int d = 0; d < 6; ++d)
        kb[d] = ld_frag2(&Kl[kt * 16 + l16][d * 32 + hi * 8], &Kl[kt * 16 + l16][d * 32 + 16 + hi * 8]);
#pragma unroll
      for (int d = 0; d < 6; ++d)
        sc[kt] = wmma_bf16(qf[d], kb[d], sc[kt]);
    }

    // ---- online softmax (rows live in 16-lane halves) ----
#pragma unroll
    for (int r = 0; r < 8; ++r) {
      float tm = fmaxf(sc[0][r], sc[1][r]);
#pragma unroll
      for (int off = 1; off < 16; off <<= 1)
        tm = fmaxf(tm, __shfl_xor(tm, off, 32));
      const float mnew  = fmaxf(mrow[r], tm);
      const float alpha = __expf(mrow[r] - mnew);
      const float p0 = __expf(sc[0][r] - mnew);
      const float p1 = __expf(sc[1][r] - mnew);
      float ts = p0 + p1;
#pragma unroll
      for (int off = 1; off < 16; off <<= 1)
        ts += __shfl_xor(ts, off, 32);
      lrow[r] = lrow[r] * alpha + ts;
      mrow[r] = mnew;
#pragma unroll
      for (int f = 0; f < 8; ++f) oacc[f][r] *= alpha;
      const int prow = hi * 8 + r;
      Pl[wave][prow][l16]      = (bf16_t)p0;
      Pl[wave][prow][16 + l16] = (bf16_t)p1;
    }
    // intra-wave LDS RAW fence (P store -> P load)
    asm volatile("s_wait_dscnt 0" ::: "memory");

    // ---- P @ V: batch fragment loads, then WMMAs ----
    {
      const bf16_t* prow = &Pl[wave][l16][0];
      const v16bf pa = ld_frag2(prow + (hi ? 8 : 0), prow + (hi ? 24 : 16));
      v16bf vb[8];
#pragma unroll
      for (int f = 0; f < 8; ++f)
        vb[f] = ld_frag2(&Vt[f * 16 + l16][hi * 8], &Vt[f * 16 + l16][16 + hi * 8]);
#pragma unroll
      for (int f = 0; f < 8; ++f)
        oacc[f] = wmma_bf16(pa, vb[f], oacc[f]);
    }
    __syncthreads();
  }

  // ---- finalize: o[s][hh*HD + d] ----
#pragma unroll
  for (int f = 0; f < 8; ++f) {
    const int col = hh * HD + f * 16 + l16;
#pragma unroll
    for (int r = 0; r < 8; ++r) {
      const int row = q0 + wave * 16 + hi * 8 + r;
      o[(size_t)row * (NH * HD) + col] = oacc[f][r] / lrow[r];
    }
  }
}

// ---------------------------------------------------------------------------
extern "C" void kernel_launch(void* const* d_in, const int* in_sizes, int n_in,
                              void* d_out, int out_size, void* d_ws, size_t ws_size,
                              hipStream_t stream)
{
  const float* h    = (const float*)d_in[0];
  const float* Wdq  = (const float*)d_in[1];
  const float* bdq  = (const float*)d_in[2];
  const float* Wdkv = (const float*)d_in[3];
  const float* bdkv = (const float*)d_in[4];
  const float* Wuq  = (const float*)d_in[5];
  const float* buq  = (const float*)d_in[6];
  const float* Wukv = (const float*)d_in[7];
  const float* bukv = (const float*)d_in[8];
  const float* Wrq  = (const float*)d_in[9];
  const float* brq  = (const float*)d_in[10];
  const float* Wrk  = (const float*)d_in[11];
  const float* brk  = (const float*)d_in[12];
  const float* Wo   = (const float*)d_in[13];
  const float* bo   = (const float*)d_in[14];
  float* out = (float*)d_out;

  const int S = in_sizes[0] / H_DIM;   // 1024

  float* ws   = (float*)d_ws;
  float* c_q  = ws;                                  // S x DQ
  float* c_kv = c_q  + (size_t)S * DQ;               // S x DKV
  float* q_cb = c_kv + (size_t)S * DKV;              // S x NH*HD
  float* kv_c = q_cb + (size_t)S * NH * HD;          // S x NH*2*HD
  float* q_rb = kv_c + (size_t)S * NH * 2 * HD;      // S x NH*RD
  float* k_rb = q_rb + (size_t)S * NH * RD;          // S x RD
  float* ob   = k_rb + (size_t)S * RD;               // S x NH*HD

  const dim3 blk(256);
  gemm_bf16<<<dim3(S / BM, DQ / BN),           blk, 0, stream>>>(h,    Wdq,  bdq,  c_q,  S, DQ,          H_DIM);
  gemm_bf16<<<dim3(S / BM, DKV / BN),          blk, 0, stream>>>(h,    Wdkv, bdkv, c_kv, S, DKV,         H_DIM);
  gemm_bf16<<<dim3(S / BM, (NH * HD) / BN),    blk, 0, stream>>>(c_q,  Wuq,  buq,  q_cb, S, NH * HD,     DQ);
  gemm_bf16<<<dim3(S / BM, (NH * 2 * HD) / BN),blk, 0, stream>>>(c_kv, Wukv, bukv, kv_c, S, NH * 2 * HD, DKV);
  gemm_bf16<<<dim3(S / BM, (NH * RD) / BN),    blk, 0, stream>>>(c_q,  Wrq,  brq,  q_rb, S, NH * RD,     DQ);
  gemm_bf16<<<dim3(S / BM, RD / BN),           blk, 0, stream>>>(h,    Wrk,  brk,  k_rb, S, RD,          H_DIM);

  {
    const int tq = S * NH * 32;
    rope_kernel<<<(tq + 255) / 256, 256, 0, stream>>>(q_rb, S, NH, NH * RD);
    const int tk = S * 32;
    rope_kernel<<<(tk + 255) / 256, 256, 0, stream>>>(k_rb, S, 1, RD);
  }

  attn_kernel<<<dim3(S / 64, NH), dim3(128), 0, stream>>>(q_cb, q_rb, kv_c, k_rb, ob, S);

  gemm_bf16<<<dim3(S / BM, H_DIM / BN), blk, 0, stream>>>(ob, Wo, bo, out, S, H_DIM, NH * HD);
}
